// InteractionPPBlock_62199716381203
// MI455X (gfx1250) — matrix operand order repaired and checked by reading, not compiled, not run on hardware
//
#include <hip/hip_runtime.h>
#include <hip/hip_bf16.h>
#include <math.h>

// ---------------------------------------------------------------------------
// Problem constants (from reference)
// ---------------------------------------------------------------------------
#define EMB        128
#define INT_EMB    64
#define NUM_RADIAL 6
#define SBF_IN     42   // NUM_RADIAL * NUM_SPHERICAL
#define ACT_STRIDE 136  // 128 + 8 bf16 pad -> row stride 272B (conflict-free A loads)

typedef __bf16 bf16_t;
typedef __attribute__((ext_vector_type(16))) __bf16 v16bf;
typedef __attribute__((ext_vector_type(8)))  __bf16 v8bf;
typedef __attribute__((ext_vector_type(8)))  float  v8f;

__device__ __forceinline__ float silu_f(float x) {
  return x / (1.0f + __expf(-x));
}

// ---------------------------------------------------------------------------
// WMMA helpers (CDNA5 wave32 layouts, ISA 7.12.2)
//  A 16x32 bf16: lane l (m=l&15, hi=l>>4) holds K = kb*32 + 8*hi + [0..7]
//                and K = kb*32 + 16 + 8*hi + [0..7]   (two 16B LDS loads)
//  B 32x16 bf16 (packed): lane l (n=l&15) holds K = 8*(l>>4) + [0..7] and 16 + 8*(l>>4) + [0..7]
//                fragment-contiguous, 32B per lane (2 x ds_load_b128)
//  C/D 16x16 f32: lane l, vgpr v -> (m = v + 8*(l>>4), n = l&15)
// ---------------------------------------------------------------------------
__device__ __forceinline__ v16bf load_a_frag(const bf16_t (*act)[ACT_STRIDE],
                                             int row, int k0) {
  v8bf lo = *(const v8bf*)&act[row][k0];
  v8bf hi = *(const v8bf*)&act[row][k0 + 16];
  return __builtin_shufflevector(lo, hi,
      0, 1, 2, 3, 4, 5, 6, 7, 8, 9, 10, 11, 12, 13, 14, 15);
}

__device__ __forceinline__ v16bf load_b_frag(const bf16_t* wp, int fragIdx, int lane) {
  return *(const v16bf*)(wp + (fragIdx * 32 + lane) * 16);
}

// Software-pipelined tile GEMM: fragment fetches run one step ahead of the
// consuming WMMA so each v_wmma overlaps the next ds_load pair.
template <int KB, int NT>
__device__ __forceinline__ void gemm_tile(const bf16_t (*act)[ACT_STRIDE],
                                          int rowBase, int lane,
                                          const bf16_t* wp,   // LDS-staged weights
                                          v8f* acc) {
  const int m   = lane & 15;
  const int hi  = lane >> 4;
  const int row = rowBase + m;

  v16bf a = load_a_frag(act, row, 8 * hi);
  v16bf b = load_b_frag(wp, 0, lane);
#pragma unroll
  for (int kb = 0; kb < KB; ++kb) {
    v16bf aNext = a;
    if (kb + 1 < KB) aNext = load_a_frag(act, row, (kb + 1) * 32 + 8 * hi);
#pragma unroll
    for (int nt = 0; nt < NT; ++nt) {
      const int idx = kb * NT + nt;
      v16bf bNext = b;
      if (idx + 1 < KB * NT) bNext = load_b_frag(wp, idx + 1, lane);
      acc[nt] = __builtin_amdgcn_wmma_f32_16x16x32_bf16(
          false, a, false, b, (short)0, acc[nt], false, false);
      b = bNext;
    }
    a = aNext;
  }
}

// Cooperative global->LDS copy of a packed weight stage (16B vectors)
__device__ __forceinline__ void copy_w(bf16_t* dst, const bf16_t* __restrict__ src,
                                       int nelems, int tid) {
  const int nvec = nelems >> 3;
  for (int i = tid; i < nvec; i += 256)
    ((v8bf*)dst)[i] = ((const v8bf*)src)[i];
}

// ---------------------------------------------------------------------------
// One-shot prep kernels
// ---------------------------------------------------------------------------
__global__ void fold_rbf_kernel(const float* __restrict__ W1,
                                const float* __restrict__ W2,
                                float* __restrict__ Wout) {
  const int idx = blockIdx.x * blockDim.x + threadIdx.x;
  if (idx < NUM_RADIAL * EMB) {
    const int r = idx / EMB, n = idx % EMB;
    float s = 0.f;
#pragma unroll
    for (int j = 0; j < 8; ++j) s += W1[r * 8 + j] * W2[j * EMB + n];
    Wout[idx] = s;
  }
}

// Pack row-major f32 weight [Ktot][Ntot] into bf16 WMMA-B fragment layout
__global__ void pack_weight_kernel(const float* __restrict__ W,
                                   bf16_t* __restrict__ dstp,
                                   int Ktot, int Ntot) {
  const int NT    = Ntot >> 4;
  const int total = Ktot * Ntot;
  for (int p = blockIdx.x * blockDim.x + threadIdx.x; p < total;
       p += gridDim.x * blockDim.x) {
    const int e  = p & 15;
    const int l  = (p >> 4) & 31;
    const int f  = p >> 9;             // fragment index = kb*NT + nt
    const int nt = f % NT;
    const int kb = f / NT;
    const int n  = nt * 16 + (l & 15);
    const int k  = kb * 32 + ((e & 7) + 8 * (l >> 4) + 16 * (e >> 3));
    dstp[p] = (bf16_t)W[k * Ntot + n];
  }
}

__global__ void zero_kernel(float* __restrict__ p, long n) {
  for (long i = (long)blockIdx.x * blockDim.x + threadIdx.x; i < n;
       i += (long)gridDim.x * blockDim.x)
    p[i] = 0.f;
}

// ---------------------------------------------------------------------------
// Edge front: x_ji = silu(m@W_ji+b); t = silu(m@W_kj+b)*rbf_e;
//             x_kj = silu(t@W_down).  128 rows/block, wave-private 16-row tiles.
// Weight stages staged through LDS (one 32KB buffer, copy overlaps epilogue).
// ---------------------------------------------------------------------------
__global__ __launch_bounds__(256)
void edge_front_kernel(const float* __restrict__ mIn,
                       const float* __restrict__ rbf,
                       const float* __restrict__ Wrbf,   // folded [6][128] f32
                       const bf16_t* __restrict__ pW_ji, const float* __restrict__ b_ji,
                       const bf16_t* __restrict__ pW_kj, const float* __restrict__ b_kj,
                       const bf16_t* __restrict__ pW_down,
                       float* __restrict__ x_ji_out,     // [E][128]
                       float* __restrict__ x_kj_out)     // [E][64]
{
  __shared__ bf16_t act[128][ACT_STRIDE];
  __shared__ bf16_t wlds[EMB * EMB];
  __shared__ float  s_rbf[128][8];

  const int  tid      = threadIdx.x;
  const int  wave     = tid >> 5;
  const int  lane     = tid & 31;
  const int  rowBase  = wave * 16;
  const long blockRow = (long)blockIdx.x * 128;
  const int  mlo      = lane & 15;
  const int  hi       = lane >> 4;

  // Stage m rows -> bf16 LDS tile; rbf rows; first weight stage
  for (int r = 0; r < 16; ++r) {
    const int    row = rowBase + r;
    const float4 v   = *(const float4*)(mIn + (blockRow + row) * EMB + lane * 4);
    bf16_t* d = &act[row][lane * 4];
    d[0] = (bf16_t)v.x; d[1] = (bf16_t)v.y; d[2] = (bf16_t)v.z; d[3] = (bf16_t)v.w;
  }
  for (int idx = lane; idx < 16 * NUM_RADIAL; idx += 32) {
    const int r = idx / NUM_RADIAL, c = idx % NUM_RADIAL;
    s_rbf[rowBase + r][c] = rbf[(blockRow + rowBase + r) * NUM_RADIAL + c];
  }
  copy_w(wlds, pW_ji, EMB * EMB, tid);
  __syncthreads();

  v8f acc[8];
  const v8f vz = {0.f, 0.f, 0.f, 0.f, 0.f, 0.f, 0.f, 0.f};

  // ---- stage 0: x_ji ----
#pragma unroll
  for (int i = 0; i < 8; ++i) acc[i] = vz;
  gemm_tile<4, 8>(act, rowBase, lane, wlds, acc);
  __syncthreads();                       // everyone done reading wlds
  copy_w(wlds, pW_kj, EMB * EMB, tid);   // prefetch next stage
#pragma unroll
  for (int nt = 0; nt < 8; ++nt) {
    const int   n = mlo + 16 * nt;
    const float b = b_ji[n];
#pragma unroll
    for (int v = 0; v < 8; ++v) {
      const long grow = blockRow + rowBase + v + 8 * hi;
      x_ji_out[grow * EMB + n] = silu_f(acc[nt][v] + b);
    }
  }
  __syncthreads();

  // ---- stage 1: x_kj * rbf_e -> back into act (own rows) ----
#pragma unroll
  for (int i = 0; i < 8; ++i) acc[i] = vz;
  gemm_tile<4, 8>(act, rowBase, lane, wlds, acc);
  __syncthreads();
  copy_w(wlds, pW_down, EMB * INT_EMB, tid);
#pragma unroll
  for (int nt = 0; nt < 8; ++nt) {
    const int   n = mlo + 16 * nt;
    const float b = b_kj[n];
    float wr[NUM_RADIAL];
#pragma unroll
    for (int r = 0; r < NUM_RADIAL; ++r) wr[r] = Wrbf[r * EMB + n];
#pragma unroll
    for (int v = 0; v < 8; ++v) {
      const int rloc = rowBase + v + 8 * hi;
      float rb = 0.f;
#pragma unroll
      for (int r = 0; r < NUM_RADIAL; ++r) rb += s_rbf[rloc][r] * wr[r];
      act[rloc][n] = (bf16_t)(silu_f(acc[nt][v] + b) * rb);
    }
  }
  __syncthreads();

  // ---- stage 2: down-projection K=128 -> N=64 ----
  v8f acc2[4];
#pragma unroll
  for (int i = 0; i < 4; ++i) acc2[i] = vz;
  gemm_tile<4, 4>(act, rowBase, lane, wlds, acc2);
#pragma unroll
  for (int nt = 0; nt < 4; ++nt) {
    const int n = mlo + 16 * nt;
#pragma unroll
    for (int v = 0; v < 8; ++v) {
      const long grow = blockRow + rowBase + v + 8 * hi;
      x_kj_out[grow * INT_EMB + n] = silu_f(acc2[nt][v]);
    }
  }
}

// ---------------------------------------------------------------------------
// Triplet side: sbf_t = (sbf@W_sbf1)@W_sbf2; msg = x_kj[src]*sbf_t;
// atomic segment-sum into aggr[dst]. 64 triplets/block, 4 threads/triplet.
// ---------------------------------------------------------------------------
__global__ __launch_bounds__(256)
void triplet_scatter_kernel(const float* __restrict__ sbf,
                            const int* __restrict__ srcI,
                            const int* __restrict__ dstI,
                            const float* __restrict__ W1,   // [42][8]
                            const float* __restrict__ W2,   // [8][64]
                            const float* __restrict__ xkj,  // [E][64]
                            float* __restrict__ accum)      // [E][64]
{
  __shared__ float s_sbf[64][44];
  __shared__ float s_h[64][8];

  const int  tid = threadIdx.x;
  const long t0  = (long)blockIdx.x * 64;

  for (int idx = tid; idx < 64 * SBF_IN; idx += 256) {
    const int r = idx / SBF_IN, c = idx % SBF_IN;
    s_sbf[r][c] = sbf[(t0 + r) * SBF_IN + c];
  }
  __syncthreads();

  for (int idx = tid; idx < 64 * 8; idx += 256) {
    const int r = idx >> 3, j = idx & 7;
    float s = 0.f;
#pragma unroll
    for (int i = 0; i < SBF_IN; ++i) s += s_sbf[r][i] * W1[i * 8 + j];
    s_h[r][j] = s;
  }
  __syncthreads();

  const int  tl   = tid >> 2;        // local triplet 0..63
  const int  part = tid & 3;         // 16-column group
  const long t    = t0 + tl;
  const int  si   = srcI[t];
  const int  di   = dstI[t];

  float h[8];
#pragma unroll
  for (int j = 0; j < 8; ++j) h[j] = s_h[tl][j];

  const float* xk = xkj + (long)si * INT_EMB + part * 16;
  float*       ac = accum + (long)di * INT_EMB + part * 16;

#pragma unroll
  for (int c = 0; c < 16; ++c) {
    const int col = part * 16 + c;
    float s = 0.f;
#pragma unroll
    for (int j = 0; j < 8; ++j) s += h[j] * W2[j * INT_EMB + col];
    const float msg = s * xk[c];
    __hip_atomic_fetch_add(&ac[c], msg, __ATOMIC_RELAXED,
                           __HIP_MEMORY_SCOPE_AGENT);
  }
}

// ---------------------------------------------------------------------------
// Edge back: fused silu(aggr@W_up)+x_ji -> residual -> final(+m) -> 2x residual
// Uniform 8-stage pipeline; each stage's weights staged via LDS, the copy for
// stage s+1 overlapping stage s's epilogue.
// Stage kinds: 0=UP(silu+x_ji)  1=MID(silu->act)  2=RES(xs+=silu->act)
//              3=FIN(m+silu->act)  4=RES_OUT(xs+silu->global out)
// ---------------------------------------------------------------------------
__global__ __launch_bounds__(256)
void edge_back_kernel(const float* __restrict__ aggr,   // [E][64]
                      const float* __restrict__ x_ji,   // [E][128]
                      const float* __restrict__ mIn,    // [E][128]
                      const bf16_t* __restrict__ pW_up,
                      const bf16_t* __restrict__ pWb1, const float* __restrict__ bb1,
                      const bf16_t* __restrict__ pWb2, const float* __restrict__ bb2,
                      const bf16_t* __restrict__ pW_fin, const float* __restrict__ b_fin,
                      const bf16_t* __restrict__ pWa1, const float* __restrict__ ba1,
                      const bf16_t* __restrict__ pWa2, const float* __restrict__ ba2,
                      float* __restrict__ outp)         // [E][128]
{
  __shared__ bf16_t act[128][ACT_STRIDE];
  __shared__ bf16_t wlds[EMB * EMB];

  const int  tid      = threadIdx.x;
  const int  wave     = tid >> 5;
  const int  lane     = tid & 31;
  const int  rowBase  = wave * 16;
  const long blockRow = (long)blockIdx.x * 128;
  const int  mlo      = lane & 15;
  const int  hi       = lane >> 4;

  const bf16_t* wsrc[8] = {pW_up, pWb1, pWb2, pW_fin,
                           pWa1, pWa2, pWa1 + EMB * EMB, pWa2 + EMB * EMB};
  const float*  bias[8] = {nullptr, bb1, bb2, b_fin, ba1, ba2, ba1 + EMB, ba2 + EMB};
  const int     kind[8] = {0, 1, 2, 3, 1, 2, 1, 4};

  // Stage aggregated messages (64 cols) -> bf16 LDS; first weight stage
  for (int r = 0; r < 16; ++r) {
    const int    row = rowBase + r;
    const float2 v   = *(const float2*)(aggr + (blockRow + row) * INT_EMB + lane * 2);
    act[row][lane * 2]     = (bf16_t)v.x;
    act[row][lane * 2 + 1] = (bf16_t)v.y;
  }
  copy_w(wlds, wsrc[0], INT_EMB * EMB, tid);
  __syncthreads();

  v8f acc[8], xs[8];
  const v8f vz = {0.f, 0.f, 0.f, 0.f, 0.f, 0.f, 0.f, 0.f};

  for (int s = 0; s < 8; ++s) {
#pragma unroll
    for (int i = 0; i < 8; ++i) acc[i] = vz;
    if (s == 0) gemm_tile<2, 8>(act, rowBase, lane, wlds, acc);
    else        gemm_tile<4, 8>(act, rowBase, lane, wlds, acc);
    __syncthreads();                       // all waves done reading wlds + act
    if (s < 7) copy_w(wlds, wsrc[s + 1], EMB * EMB, tid);  // prefetch next stage

    const int k = kind[s];
#pragma unroll
    for (int nt = 0; nt < 8; ++nt) {
      const int   n = mlo + 16 * nt;
      const float b = (k == 0) ? 0.f : bias[s][n];
#pragma unroll
      for (int v = 0; v < 8; ++v) {
        const long grow = blockRow + rowBase + v + 8 * hi;
        const int  rloc = rowBase + v + 8 * hi;
        const float sv  = silu_f(acc[nt][v] + b);
        if (k == 0) {
          const float val = sv + x_ji[grow * EMB + n];
          xs[nt][v] = val;
          act[rloc][n] = (bf16_t)val;
        } else if (k == 1) {
          act[rloc][n] = (bf16_t)sv;
        } else if (k == 2) {
          const float val = xs[nt][v] + sv;
          xs[nt][v] = val;
          act[rloc][n] = (bf16_t)val;
        } else if (k == 3) {
          const float val = mIn[grow * EMB + n] + sv;
          xs[nt][v] = val;
          act[rloc][n] = (bf16_t)val;
        } else {
          outp[grow * EMB + n] = xs[nt][v] + sv;
        }
      }
    }
    __syncthreads();
  }
}

// ---------------------------------------------------------------------------
// Launcher
// ---------------------------------------------------------------------------
extern "C" void kernel_launch(void* const* d_in, const int* in_sizes, int n_in,
                              void* d_out, int out_size, void* d_ws, size_t ws_size,
                              hipStream_t stream) {
  (void)n_in; (void)out_size; (void)ws_size;

  const float* m      = (const float*)d_in[0];
  const float* rbf    = (const float*)d_in[1];
  const float* sbf    = (const float*)d_in[2];
  const int*   srcI   = (const int*)d_in[3];
  const int*   dstI   = (const int*)d_in[4];
  const float* W_rbf1 = (const float*)d_in[5];
  const float* W_rbf2 = (const float*)d_in[6];
  const float* W_sbf1 = (const float*)d_in[7];
  const float* W_sbf2 = (const float*)d_in[8];
  const float* W_ji   = (const float*)d_in[9];
  const float* b_ji   = (const float*)d_in[10];
  const float* W_kj   = (const float*)d_in[11];
  const float* b_kj   = (const float*)d_in[12];
  const float* W_down = (const float*)d_in[13];
  const float* W_up   = (const float*)d_in[14];
  const float* Wb1    = (const float*)d_in[15];
  const float* bb1    = (const float*)d_in[16];
  const float* Wb2    = (const float*)d_in[17];
  const float* bb2    = (const float*)d_in[18];
  const float* W_fin  = (const float*)d_in[19];
  const float* b_fin  = (const float*)d_in[20];
  const float* Wa1    = (const float*)d_in[21];
  const float* ba1    = (const float*)d_in[22];
  const float* Wa2    = (const float*)d_in[23];
  const float* ba2    = (const float*)d_in[24];

  const long E = in_sizes[0] / EMB;   // 262144
  const long T = in_sizes[3];         // 2097152

  // Workspace carve-out
  char*  ws  = (char*)d_ws;
  size_t off = 0;
  auto carve = [&](size_t bytes) -> void* {
    void* p = ws + off;
    off = (off + bytes + 255) & ~(size_t)255;
    return p;
  };
  float*  x_ji   = (float*)carve((size_t)E * EMB * sizeof(float));
  float*  x_kj   = (float*)carve((size_t)E * INT_EMB * sizeof(float));
  float*  aggr   = (float*)carve((size_t)E * INT_EMB * sizeof(float));
  float*  Wrbf   = (float*)carve((size_t)NUM_RADIAL * EMB * sizeof(float));
  bf16_t* pW_ji  = (bf16_t*)carve(16384 * sizeof(bf16_t));
  bf16_t* pW_kj  = (bf16_t*)carve(16384 * sizeof(bf16_t));
  bf16_t* pW_dn  = (bf16_t*)carve(8192 * sizeof(bf16_t));
  bf16_t* pW_up  = (bf16_t*)carve(8192 * sizeof(bf16_t));
  bf16_t* pWb1   = (bf16_t*)carve(16384 * sizeof(bf16_t));
  bf16_t* pWb2   = (bf16_t*)carve(16384 * sizeof(bf16_t));
  bf16_t* pW_fin = (bf16_t*)carve(16384 * sizeof(bf16_t));
  bf16_t* pWa1   = (bf16_t*)carve(2 * 16384 * sizeof(bf16_t));
  bf16_t* pWa2   = (bf16_t*)carve(2 * 16384 * sizeof(bf16_t));

  // --- prep: fold rbf basis, pack all WMMA weights to bf16 fragments ---
  fold_rbf_kernel<<<3, 256, 0, stream>>>(W_rbf1, W_rbf2, Wrbf);
  pack_weight_kernel<<<64, 256, 0, stream>>>(W_ji,  pW_ji,  128, 128);
  pack_weight_kernel<<<64, 256, 0, stream>>>(W_kj,  pW_kj,  128, 128);
  pack_weight_kernel<<<32, 256, 0, stream>>>(W_down, pW_dn, 128, 64);
  pack_weight_kernel<<<32, 256, 0, stream>>>(W_up,  pW_up,  64, 128);
  pack_weight_kernel<<<64, 256, 0, stream>>>(Wb1,   pWb1,   128, 128);
  pack_weight_kernel<<<64, 256, 0, stream>>>(Wb2,   pWb2,   128, 128);
  pack_weight_kernel<<<64, 256, 0, stream>>>(W_fin, pW_fin, 128, 128);
  pack_weight_kernel<<<64, 256, 0, stream>>>(Wa1,          pWa1,          128, 128);
  pack_weight_kernel<<<64, 256, 0, stream>>>(Wa1 + 16384,  pWa1 + 16384,  128, 128);
  pack_weight_kernel<<<64, 256, 0, stream>>>(Wa2,          pWa2,          128, 128);
  pack_weight_kernel<<<64, 256, 0, stream>>>(Wa2 + 16384,  pWa2 + 16384,  128, 128);

  // --- zero the segment-sum accumulator ---
  zero_kernel<<<2048, 256, 0, stream>>>(aggr, E * INT_EMB);

  // --- edge front (WMMA chain, LDS-staged weights) ---
  edge_front_kernel<<<(int)(E / 128), 256, 0, stream>>>(
      m, rbf, Wrbf, pW_ji, b_ji, pW_kj, b_kj, pW_dn, x_ji, x_kj);

  // --- triplet gather/scatter segment-sum ---
  triplet_scatter_kernel<<<(int)(T / 64), 256, 0, stream>>>(
      sbf, srcI, dstI, W_sbf1, W_sbf2, x_kj, aggr);

  // --- edge back (fused 8-stage WMMA pipeline, LDS-staged weights) ---
  edge_back_kernel<<<(int)(E / 128), 256, 0, stream>>>(
      aggr, x_ji, m, pW_up, pWb1, bb1, pWb2, bb2, pW_fin, b_fin,
      pWa1, ba1, pWa2, ba2, (float*)d_out);
}
